// GCN_8632884265528
// MI455X (gfx1250) — matrix operand order repaired and checked by reading, not compiled, not run on hardware
//
#include <hip/hip_runtime.h>
#include <hip/hip_bf16.h>

typedef __attribute__((ext_vector_type(16))) _Float16 v16h;
typedef __attribute__((ext_vector_type(8)))  float    v8f;

#define N_NODES 50000
#define N_EDGES 800000
#define IN_F    128
#define HID_F   128
#define OUT_F   64

// ---------------------------------------------------------------------------
// Weight prep: Wt[n][k] = (f16) W[k][n]   (K-major f16 so B fragments are
// contiguous 32B runs -> global_load_b128 instead of 16 strided b32 loads)
// ---------------------------------------------------------------------------
__global__ void gcn_w_transpose_f16(const float* __restrict__ W,
                                    _Float16* __restrict__ Wt,
                                    int K, int N)
{
    int t = blockIdx.x * blockDim.x + threadIdx.x;
    if (t >= K * N) return;
    int k = t / N;
    int n = t - k * N;
    Wt[(size_t)n * K + k] = (_Float16)W[(size_t)k * N + n];
}

// ---------------------------------------------------------------------------
// WMMA GEMM: out[M,N] = (relu? max(A,0):A)[M,K] @ W[K,N], f16 in, f32 acc.
// Wt is the transposed f16 weight [N][K]. One wave owns a 16x64 tile.
// A-fragment (16-bit A 16x32): lane = 16*half + m
//   a[i]   = A[m][kb + 16*half + i]   (i = 0..15, contiguous k)
// B-fragment (16-bit B 32x16):
//   b[i] = W[kb + 16*half + i][n] = Wt[n][kb + 16*half + i]  -> one v16h load
// C/D (f32 16x16): VGPR j -> row (j + 8*half), col (n0 + m)
// ---------------------------------------------------------------------------
__global__ void gcn_gemm_wmma_f16(const float* __restrict__ A,
                                  const _Float16* __restrict__ Wt,
                                  float* __restrict__ out,
                                  int M, int K, int N, int relu)
{
    const int lane  = threadIdx.x & 31;
    const int wave  = threadIdx.x >> 5;      // 8 waves / block
    const int half_ = lane >> 4;             // 0 or 1
    const int m16   = lane & 15;

    const int rowTile = blockIdx.x * 8 + wave;
    const int row0    = rowTile * 16;
    const int n0      = blockIdx.y * 64;
    if (row0 >= M) return;                   // wave-uniform: EXEC stays all-1s

    v8f acc[4] = {v8f{}, v8f{}, v8f{}, v8f{}};

    for (int kb = 0; kb < K; kb += 32) {
        // ---- A fragment: one 64B run, vectorized float4 loads ----
        const float4* ap4 =
            (const float4*)(A + (size_t)(row0 + m16) * K + kb + 16 * half_);
        float4 q0 = ap4[0];          // A[m][kb+16h .. +3]
        float4 q1 = ap4[1];          // A[m][kb+16h+4 .. +7]
        float4 q2 = ap4[2];          // A[m][kb+16h+8 .. +11]
        float4 q3 = ap4[3];          // A[m][kb+16h+12 .. +15]
        if (relu) {
            q0.x=fmaxf(q0.x,0.f); q0.y=fmaxf(q0.y,0.f); q0.z=fmaxf(q0.z,0.f); q0.w=fmaxf(q0.w,0.f);
            q1.x=fmaxf(q1.x,0.f); q1.y=fmaxf(q1.y,0.f); q1.z=fmaxf(q1.z,0.f); q1.w=fmaxf(q1.w,0.f);
            q2.x=fmaxf(q2.x,0.f); q2.y=fmaxf(q2.y,0.f); q2.z=fmaxf(q2.z,0.f); q2.w=fmaxf(q2.w,0.f);
            q3.x=fmaxf(q3.x,0.f); q3.y=fmaxf(q3.y,0.f); q3.z=fmaxf(q3.z,0.f); q3.w=fmaxf(q3.w,0.f);
        }
        v16h a;
        a[0]=(_Float16)q0.x;  a[1]=(_Float16)q0.y;  a[2]=(_Float16)q0.z;  a[3]=(_Float16)q0.w;
        a[4]=(_Float16)q1.x;  a[5]=(_Float16)q1.y;  a[6]=(_Float16)q1.z;  a[7]=(_Float16)q1.w;
        a[8]=(_Float16)q2.x;  a[9]=(_Float16)q2.y;  a[10]=(_Float16)q2.z; a[11]=(_Float16)q2.w;
        a[12]=(_Float16)q3.x; a[13]=(_Float16)q3.y; a[14]=(_Float16)q3.z; a[15]=(_Float16)q3.w;

        // ---- 4 column tiles of 16: each B fragment is one 32B v16h load ----
#pragma unroll
        for (int t = 0; t < 4; ++t) {
            const int n = n0 + t * 16 + m16;
            v16h b = *(const v16h*)(Wt + (size_t)n * K + kb + 16 * half_);
            acc[t] = __builtin_amdgcn_wmma_f32_16x16x32_f16(
                /*neg_a=*/false, a, /*neg_b=*/false, b,
                /*c_mod=*/(short)0, acc[t],
                /*reuse_a=*/false, /*reuse_b=*/false);
        }
    }

    // ---- store D per the f32 C/D VGPR layout ----
#pragma unroll
    for (int t = 0; t < 4; ++t) {
        float* op = out + (size_t)row0 * N + n0 + t * 16 + m16;
#pragma unroll
        for (int j = 0; j < 8; ++j)
            op[(size_t)(j + 8 * half_) * N] = acc[t][j];
    }
}

// ---------------------------------------------------------------------------
// dst[i, f] = bias[f]  (accumulator init: bias folded in before scatter-add)
// ---------------------------------------------------------------------------
__global__ void gcn_bias_init(float* __restrict__ dst,
                              const float* __restrict__ bias,
                              int total, int F)
{
    int t = blockIdx.x * blockDim.x + threadIdx.x;
    if (t >= total) return;
    dst[t] = bias[t % F];
}

// ---------------------------------------------------------------------------
// Edge scatter: dst[row[e], :] += val[e] * src[col[e], :]
// thread t handles one float4 chunk; for F=128 one wave == one 512B row.
// src (25.6 / 12.8 MB) is L2-resident, so gathers + atomics resolve in L2.
// ---------------------------------------------------------------------------
__global__ void gcn_spmm_scatter(const int* __restrict__ erow,
                                 const int* __restrict__ ecol,
                                 const float* __restrict__ eval,
                                 const float* __restrict__ src,
                                 float* __restrict__ dst,
                                 int E, int F)
{
    const int n4 = F >> 2;
    int t = blockIdx.x * blockDim.x + threadIdx.x;
    int e = t / n4;
    if (e >= E) return;
    int i = t - e * n4;

    const int   r = erow[e];
    const int   c = ecol[e];
    const float v = eval[e];

    const float4 x = ((const float4*)(src + (size_t)c * F))[i];
    float* d = dst + (size_t)r * F + (size_t)i * 4;
    atomicAdd(d + 0, v * x.x);
    atomicAdd(d + 1, v * x.y);
    atomicAdd(d + 2, v * x.z);
    atomicAdd(d + 3, v * x.w);
}

// ---------------------------------------------------------------------------
// Launch: Wt prep ; t1 = x@W1 ; h = b1 ; h += scatter(t1) ;
//         t2 = relu(h)@W2 ; out = b2 ; out += scatter(t2)
// ---------------------------------------------------------------------------
extern "C" void kernel_launch(void* const* d_in, const int* in_sizes, int n_in,
                              void* d_out, int out_size, void* d_ws, size_t ws_size,
                              hipStream_t stream)
{
    const float* x     = (const float*)d_in[0];
    const int*   erow  = (const int*)  d_in[1];
    const int*   ecol  = (const int*)  d_in[2];
    const float* eval  = (const float*)d_in[3];
    const float* W1    = (const float*)d_in[4];
    const float* b1    = (const float*)d_in[5];
    const float* W2    = (const float*)d_in[6];
    const float* b2    = (const float*)d_in[7];
    float* out = (float*)d_out;

    float*    t1  = (float*)d_ws;                          // [N_NODES, HID_F] f32
    float*    h   = t1 + (size_t)N_NODES * HID_F;          // [N_NODES, HID_F] f32
    float*    t2  = h  + (size_t)N_NODES * HID_F;          // [N_NODES, OUT_F] f32
    _Float16* w1t = (_Float16*)(t2 + (size_t)N_NODES * OUT_F); // [HID_F][IN_F] f16
    _Float16* w2t = w1t + (size_t)HID_F * IN_F;                // [OUT_F][HID_F] f16

    const int rowTiles   = N_NODES / 16;               // 3125 (exact)
    const int gemmBlocks = (rowTiles + 7) / 8;         // 8 waves per block

    // 0) transpose + f16-convert weights (tiny, one-time per call)
    gcn_w_transpose_f16<<<(IN_F * HID_F + 255) / 256, 256, 0, stream>>>(
        W1, w1t, IN_F, HID_F);
    gcn_w_transpose_f16<<<(HID_F * OUT_F + 255) / 256, 256, 0, stream>>>(
        W2, w2t, HID_F, OUT_F);

    // 1) t1 = x @ W1   (WMMA f16->f32)
    gcn_gemm_wmma_f16<<<dim3(gemmBlocks, HID_F / 64), 256, 0, stream>>>(
        x, w1t, t1, N_NODES, IN_F, HID_F, /*relu=*/0);

    // 2) h = broadcast(b1)
    {
        int total = N_NODES * HID_F;
        gcn_bias_init<<<(total + 255) / 256, 256, 0, stream>>>(h, b1, total, HID_F);
    }

    // 3) h += scatter-add over edges of t1
    {
        long long work = (long long)N_EDGES * (HID_F / 4);
        gcn_spmm_scatter<<<(int)((work + 255) / 256), 256, 0, stream>>>(
            erow, ecol, eval, t1, h, N_EDGES, HID_F);
    }

    // 4) t2 = relu(h) @ W2   (WMMA, ReLU fused into A load)
    gcn_gemm_wmma_f16<<<dim3(gemmBlocks, OUT_F / 64), 256, 0, stream>>>(
        h, w2t, t2, N_NODES, HID_F, OUT_F, /*relu=*/1);

    // 5) out = broadcast(b2)
    {
        int total = N_NODES * OUT_F;
        gcn_bias_init<<<(total + 255) / 256, 256, 0, stream>>>(out, b2, total, OUT_F);
    }

    // 6) out += scatter-add over edges of t2
    {
        long long work = (long long)N_EDGES * (OUT_F / 4);
        gcn_spmm_scatter<<<(int)((work + 255) / 256), 256, 0, stream>>>(
            erow, ecol, eval, t2, out, N_EDGES, OUT_F);
    }
}
